// MPiNetsPointNet_74629351735875
// MI455X (gfx1250) — matrix operand order repaired and checked by reading, not compiled, not run on hardware
//
#include <hip/hip_runtime.h>
#include <hip/hip_bf16.h>

typedef __attribute__((ext_vector_type(16))) _Float16 v16h;
typedef __attribute__((ext_vector_type(8)))  _Float16 v8h;
typedef __attribute__((ext_vector_type(8)))  float    v8f;

// Swizzled LDS activation layout: element (k,n) at ((k>>5)*NT + n)*32 + (k&31).
// -> B-fragment per lane = 32 contiguous bytes; store per lane = 16 contiguous bytes.

// ---------------------------------------------------------------------------
// Weight convert f32 -> f16 with zero padding of K up to Kpad (multiple of 32)
// ---------------------------------------------------------------------------
__global__ __launch_bounds__(256) void cvt_kernel(const float* __restrict__ src,
                                                  _Float16* __restrict__ dst,
                                                  int O, int K, int Kpad) {
    size_t i = (size_t)blockIdx.x * 256 + threadIdx.x;
    if (i >= (size_t)O * Kpad) return;
    int o = (int)(i / Kpad), k = (int)(i % Kpad);
    dst[i] = (k < K) ? (_Float16)src[(size_t)o * K + k] : (_Float16)0.f;
}

// ---------------------------------------------------------------------------
// Farthest point sampling: one block per batch. Emits idx + gathered coords.
// Matches jax reference: sequence starts at index 0; argmax tie -> lowest idx.
// ---------------------------------------------------------------------------
__global__ __launch_bounds__(256) void fps_kernel(const float* __restrict__ pts,
                                                  int stride, int n, int npoint,
                                                  int* __restrict__ out_idx,
                                                  float* __restrict__ out_xyz) {
    int b = blockIdx.x, tid = threadIdx.x;
    const float* P = pts + (size_t)b * n * stride;
    __shared__ float sd[256];
    __shared__ int   si[256];
    float dist[16];
    int ppt = (n + 255) >> 8;           // 16 for n=4096, 2 for n=512
    #pragma unroll
    for (int j = 0; j < 16; ++j) dist[j] = 1e10f;
    int far = 0;
    for (int it = 0; it < npoint; ++it) {
        float cx = P[(size_t)far * stride + 0];
        float cy = P[(size_t)far * stride + 1];
        float cz = P[(size_t)far * stride + 2];
        if (tid == 0) {
            out_idx[(size_t)b * npoint + it] = far;
            out_xyz[((size_t)b * npoint + it) * 3 + 0] = cx;
            out_xyz[((size_t)b * npoint + it) * 3 + 1] = cy;
            out_xyz[((size_t)b * npoint + it) * 3 + 2] = cz;
        }
        float best = -1.f; int bi = 0;
        for (int j = 0; j < ppt; ++j) {
            int p = tid * ppt + j;
            if (p < n) {
                float dx = P[(size_t)p * stride + 0] - cx;
                float dy = P[(size_t)p * stride + 1] - cy;
                float dz = P[(size_t)p * stride + 2] - cz;
                float d = dx * dx + dy * dy + dz * dz;
                if (d < dist[j]) dist[j] = d;
                if (dist[j] > best) { best = dist[j]; bi = p; }
            }
        }
        sd[tid] = best; si[tid] = bi;
        __syncthreads();
        for (int s2 = 128; s2 > 0; s2 >>= 1) {
            if (tid < s2) {
                float dv = sd[tid + s2]; int iv = si[tid + s2];
                if (dv > sd[tid] || (dv == sd[tid] && iv < si[tid])) {
                    sd[tid] = dv; si[tid] = iv;
                }
            }
            __syncthreads();
        }
        far = si[0];
        __syncthreads();
    }
}

// ---------------------------------------------------------------------------
// Ball query: one wave32 per center; ordered append via ballot/popcount.
// Pads tail with first in-radius index (center itself always qualifies).
// ---------------------------------------------------------------------------
__global__ __launch_bounds__(256) void bq_kernel(const float* __restrict__ pts,
                                                 int stride, int n,
                                                 const float* __restrict__ ctr,
                                                 int S, float r2, int nsample,
                                                 int* __restrict__ gidx) {
    int lane = threadIdx.x & 31, wave = threadIdx.x >> 5;
    int cid = blockIdx.x * 8 + wave;
    int b = cid / S, s = cid % S;
    const float* P = pts + (size_t)b * n * stride;
    float cx = ctr[((size_t)b * S + s) * 3 + 0];
    float cy = ctr[((size_t)b * S + s) * 3 + 1];
    float cz = ctr[((size_t)b * S + s) * 3 + 2];
    int* g = gidx + ((size_t)b * S + s) * nsample;
    int cnt = 0, first = -1;
    for (int base = 0; base < n && cnt < nsample; base += 32) {
        int p = base + lane;
        bool in = false;
        if (p < n) {
            float dx = P[(size_t)p * stride + 0] - cx;
            float dy = P[(size_t)p * stride + 1] - cy;
            float dz = P[(size_t)p * stride + 2] - cz;
            in = (dx * dx + dy * dy + dz * dz) < r2;
        }
        unsigned mm = (unsigned)__ballot((int)in);
        if (first < 0 && mm) first = base + __ffs((int)mm) - 1;
        if (in) {
            int pos = cnt + __popc(mm & ((1u << lane) - 1u));
            if (pos < nsample) g[pos] = p;
        }
        cnt += __popc(mm);
    }
    if (cnt > nsample) cnt = nsample;
    if (first < 0) first = 0;
    for (int j = cnt + lane; j < nsample; j += 32) g[j] = first;
}

// ---------------------------------------------------------------------------
// WMMA fragment helpers (v_wmma_f32_16x16x32_f16; A and B share this code's contiguous k order).
// A 16x32 f16 (global, row-major [M][Kpad]): lane row = lane%16;
//   halves 0..15 = K[koff..koff+15], koff = (lane>=16)?16:0.
// B 32x16 f16 from swizzled LDS: one aligned 32-byte load per lane.
// C/D 16x16 f32: lane n = lane%16; vgpr r -> M = r + ((lane>=16)?8:0).
// ---------------------------------------------------------------------------
__device__ __forceinline__ v16h load_afrag(const _Float16* __restrict__ W,
                                           int ld, int lane) {
    int row  = lane & 15;
    int koff = (lane & 16) ? 16 : 0;
    const _Float16* p = W + (size_t)row * ld + koff;
    v16h a;
    #pragma unroll
    for (int h = 0; h < 16; ++h) { a[h] = p[h]; }
    return a;
}

__device__ __forceinline__ v16h load_bfrag_sw(const _Float16* X, int NT,
                                              int kc, int no, int lane) {
    int col = no + (lane & 15);
    int kb  = (lane & 16) ? 16 : 0;
    return *(const v16h*)(X + ((size_t)(kc >> 5) * NT + col) * 32 + kb);
}

// One MLP layer: Ys = relu(W[M][Kpad] * Xs + bias); swizzled f16 in/out in LDS.
__device__ __forceinline__ void mlp_layer(const _Float16* __restrict__ W,
                                          const float* __restrict__ bias,
                                          const _Float16* Xs, _Float16* Ys,
                                          int M, int Kpad, int NT, int tid) {
    int wave = tid >> 5, lane = tid & 31;
    int nMt = M >> 4, nNt = NT >> 4, nT = nMt * nNt;
    for (int t = wave; t < nT; t += 8) {
        int mo = (t / nNt) << 4, no = (t % nNt) << 4;
        v8f acc = {};
        for (int kc = 0; kc < Kpad; kc += 32) {
            v16h a  = load_afrag(W + (size_t)mo * Kpad + kc, Kpad, lane);
            v16h bf = load_bfrag_sw(Xs, NT, kc, no, lane);
            acc = __builtin_amdgcn_wmma_f32_16x16x32_f16(false, a, false, bf,
                                                         (short)0, acc, false, false);
        }
        int n  = no + (lane & 15);
        int mb = mo + ((lane & 16) ? 8 : 0);
        v8h y;
        #pragma unroll
        for (int r = 0; r < 8; ++r) {
            float v = acc[r] + bias[mb + r];
            y[r] = (_Float16)(v > 0.f ? v : 0.f);
        }
        *(v8h*)(Ys + ((size_t)(mb >> 5) * NT + n) * 32 + (mb & 31)) = y;
    }
}

// Final layer fused with relu + max over the NT samples; writes f32 to global.
__device__ __forceinline__ void mlp_layer_maxpool(const _Float16* __restrict__ W,
                                                  const float* __restrict__ bias,
                                                  const _Float16* Xs,
                                                  int M, int Kpad, int NT, int tid,
                                                  float* __restrict__ out,
                                                  int out_stride) {
    int wave = tid >> 5, lane = tid & 31;
    int nMt = M >> 4, nNt = NT >> 4;
    int mhalf = (lane & 16) ? 8 : 0;
    for (int mt = wave; mt < nMt; mt += 8) {
        int mo = mt << 4;
        float vmax[8];
        #pragma unroll
        for (int r = 0; r < 8; ++r) vmax[r] = 0.f;  // relu outputs are >= 0
        for (int nt = 0; nt < nNt; ++nt) {
            int no = nt << 4;
            v8f acc = {};
            for (int kc = 0; kc < Kpad; kc += 32) {
                v16h a  = load_afrag(W + (size_t)mo * Kpad + kc, Kpad, lane);
                v16h bf = load_bfrag_sw(Xs, NT, kc, no, lane);
                acc = __builtin_amdgcn_wmma_f32_16x16x32_f16(false, a, false, bf,
                                                             (short)0, acc, false, false);
            }
            #pragma unroll
            for (int r = 0; r < 8; ++r) {
                float v = acc[r] + bias[mo + mhalf + r];
                v = v > 0.f ? v : 0.f;
                vmax[r] = fmaxf(vmax[r], v);
            }
        }
        // reduce across the 16 lanes holding the same M rows (offs 1,2,4,8)
        #pragma unroll
        for (int r = 0; r < 8; ++r)
            for (int off = 1; off < 16; off <<= 1)
                vmax[r] = fmaxf(vmax[r], __shfl_xor(vmax[r], off, 32));
        if ((lane & 15) == 0) {
            int mb = mo + mhalf;
            #pragma unroll
            for (int r = 0; r < 8; ++r)
                out[(size_t)(mb + r) * out_stride] = vmax[r];
        }
    }
}

// Zero the K-tail (k >= Kreal) of a swizzled activation buffer.
__device__ __forceinline__ void zero_pad_sw(_Float16* buf, int Kreal, int Kpad,
                                            int NT, int tid) {
    int tot = (Kpad >> 5) * NT * 32;
    int cstride = NT * 32;
    for (int i = tid; i < tot; i += 256) {
        int k = ((i / cstride) << 5) | (i & 31);
        if (k >= Kreal) buf[i] = (_Float16)0.f;
    }
}

// ---------------------------------------------------------------------------
// SA1: block per (s,b). Channels [dx,dy,dz,mask] pad->32; 32->64->64->64; max128.
// ---------------------------------------------------------------------------
__global__ __launch_bounds__(256) void sa1_kernel(const float* __restrict__ pc,
                                                  const int* __restrict__ gidx,
                                                  const float* __restrict__ ctr,
                                                  const _Float16* W1, const float* b1,
                                                  const _Float16* W2, const float* b2,
                                                  const _Float16* W3, const float* b3,
                                                  float* __restrict__ feats_out) {
    __shared__ __align__(32) _Float16 bufA[64 * 128];
    __shared__ __align__(32) _Float16 bufB[64 * 128];
    int s = blockIdx.x, b = blockIdx.y, tid = threadIdx.x;
    const int* g = gidx + ((size_t)b * 512 + s) * 128;
    zero_pad_sw(bufA, 4, 32, 128, tid);
    float cx = ctr[((size_t)b * 512 + s) * 3 + 0];
    float cy = ctr[((size_t)b * 512 + s) * 3 + 1];
    float cz = ctr[((size_t)b * 512 + s) * 3 + 2];
    if (tid < 128) {
        int p = g[tid];
        const float* q = pc + ((size_t)b * 4096 + p) * 4;
        // swizzled (k,n): n*32 + k   (chunk 0)
        bufA[tid * 32 + 0] = (_Float16)(q[0] - cx);
        bufA[tid * 32 + 1] = (_Float16)(q[1] - cy);
        bufA[tid * 32 + 2] = (_Float16)(q[2] - cz);
        bufA[tid * 32 + 3] = (_Float16)q[3];
    }
    __syncthreads();
    mlp_layer(W1, b1, bufA, bufB, 64, 32, 128, tid);
    __syncthreads();
    mlp_layer(W2, b2, bufB, bufA, 64, 64, 128, tid);
    __syncthreads();
    mlp_layer_maxpool(W3, b3, bufA, 64, 64, 128, tid,
                      feats_out + (size_t)b * 64 * 512 + s, 512);
}

// ---------------------------------------------------------------------------
// SA2: block per (s,b). Channels [dxyz, 64 feats] pad->96; ->128->128->256; max128.
// ---------------------------------------------------------------------------
__global__ __launch_bounds__(256) void sa2_kernel(const float* __restrict__ xyz,
                                                  const float* __restrict__ feats,
                                                  const int* __restrict__ gidx,
                                                  const float* __restrict__ ctr,
                                                  const _Float16* W1, const float* b1,
                                                  const _Float16* W2, const float* b2,
                                                  const _Float16* W3, const float* b3,
                                                  float* __restrict__ feats_out) {
    __shared__ __align__(32) _Float16 bufA[128 * 128];
    __shared__ __align__(32) _Float16 bufB[128 * 128];
    __shared__ int sidx[128];
    int s = blockIdx.x, b = blockIdx.y, tid = threadIdx.x;
    const int* g = gidx + ((size_t)b * 128 + s) * 128;
    if (tid < 128) sidx[tid] = g[tid];
    zero_pad_sw(bufA, 67, 96, 128, tid);
    __syncthreads();
    float cx = ctr[((size_t)b * 128 + s) * 3 + 0];
    float cy = ctr[((size_t)b * 128 + s) * 3 + 1];
    float cz = ctr[((size_t)b * 128 + s) * 3 + 2];
    if (tid < 128) {
        int p = sidx[tid];
        const float* q = xyz + ((size_t)b * 512 + p) * 3;
        bufA[tid * 32 + 0] = (_Float16)(q[0] - cx);
        bufA[tid * 32 + 1] = (_Float16)(q[1] - cy);
        bufA[tid * 32 + 2] = (_Float16)(q[2] - cz);
    }
    for (int i = tid; i < 64 * 128; i += 256) {
        int c = i >> 7, n = i & 127;
        int k = 3 + c;
        int p = sidx[n];
        bufA[((size_t)(k >> 5) * 128 + n) * 32 + (k & 31)] =
            (_Float16)feats[((size_t)b * 64 + c) * 512 + p];
    }
    __syncthreads();
    mlp_layer(W1, b1, bufA, bufB, 128, 96, 128, tid);
    __syncthreads();
    mlp_layer(W2, b2, bufB, bufA, 128, 128, 128, tid);
    __syncthreads();
    mlp_layer_maxpool(W3, b3, bufA, 256, 128, 128, tid,
                      feats_out + (size_t)b * 256 * 128 + s, 128);
}

// ---------------------------------------------------------------------------
// SA3 (group all): block per (quarter,b), 32 pixels each. 259 pad->288;
// ->512->512->1024; partial max over 32 pixels -> part[q][b][1024].
// ---------------------------------------------------------------------------
__global__ __launch_bounds__(256) void sa3_kernel(const float* __restrict__ xyz,
                                                  const float* __restrict__ feats,
                                                  const _Float16* W1, const float* b1,
                                                  const _Float16* W2, const float* b2,
                                                  const _Float16* W3, const float* b3,
                                                  float* __restrict__ part) {
    __shared__ __align__(32) _Float16 buf0[512 * 32];
    __shared__ __align__(32) _Float16 buf1[512 * 32];
    int q = blockIdx.x, b = blockIdx.y, tid = threadIdx.x;
    zero_pad_sw(buf0, 259, 288, 32, tid);
    for (int i = tid; i < 3 * 32; i += 256) {
        int c = i >> 5, n = i & 31;
        int k = q * 32 + n;
        buf0[n * 32 + c] = (_Float16)xyz[((size_t)b * 128 + k) * 3 + c];
    }
    for (int i = tid; i < 256 * 32; i += 256) {
        int c = i >> 5, n = i & 31;
        int k = 3 + c;
        buf0[((size_t)(k >> 5) * 32 + n) * 32 + (k & 31)] =
            (_Float16)feats[(size_t)b * 256 * 128 + (size_t)c * 128 + q * 32 + n];
    }
    __syncthreads();
    mlp_layer(W1, b1, buf0, buf1, 512, 288, 32, tid);
    __syncthreads();
    mlp_layer(W2, b2, buf1, buf0, 512, 512, 32, tid);
    __syncthreads();
    mlp_layer_maxpool(W3, b3, buf0, 1024, 512, 32, tid,
                      part + ((size_t)q * 16 + b) * 1024, 1);
}

__global__ __launch_bounds__(256) void maxcomb_kernel(const float* __restrict__ part,
                                                      float* __restrict__ h,
                                                      int n, int parts) {
    int i = blockIdx.x * 256 + threadIdx.x;
    if (i < n) {
        float m = part[i];
        for (int qq = 1; qq < parts; ++qq) m = fmaxf(m, part[(size_t)qq * n + i]);
        h[i] = m;
    }
}

// ---------------------------------------------------------------------------
// FC GEMM: out[16][O] = X[16][K] @ W[O][K]^T + bias. One wave per 16-out tile.
// ---------------------------------------------------------------------------
__global__ __launch_bounds__(256) void fc_kernel(const float* __restrict__ X,
                                                 const _Float16* __restrict__ W,
                                                 const float* __restrict__ bias,
                                                 float* __restrict__ out,
                                                 int K, int O) {
    int tid = threadIdx.x, wave = tid >> 5, lane = tid & 31;
    int o0 = (blockIdx.x * 8 + wave) * 16;
    if (o0 >= O) return;
    int row  = lane & 15;
    int aoff = (lane & 16) ? 16 : 0;
    int col  = lane & 15;
    int kb   = (lane & 16) ? 16 : 0;
    v8f acc = {};
    for (int kc = 0; kc < K; kc += 32) {
        v16h a, bf;
        const float* pa = X + (size_t)row * K + kc + aoff;
        #pragma unroll
        for (int h = 0; h < 16; ++h) {
            a[h]     = (_Float16)pa[h];
        }
        const _Float16* pb = W + (size_t)(o0 + col) * K + kc + kb;
        #pragma unroll
        for (int h = 0; h < 16; ++h) bf[h] = pb[h];
        acc = __builtin_amdgcn_wmma_f32_16x16x32_f16(false, a, false, bf,
                                                     (short)0, acc, false, false);
    }
    int o  = o0 + col;
    int mb = (lane & 16) ? 8 : 0;
    #pragma unroll
    for (int r = 0; r < 8; ++r)
        out[(size_t)(mb + r) * O + o] = acc[r] + bias[o];
}

// GroupNorm(16 groups) + leaky_relu(0.01), in-place. grid = (16 groups, 16 batch).
__global__ __launch_bounds__(256) void gn_kernel(float* __restrict__ x,
                                                 const float* __restrict__ gamma,
                                                 const float* __restrict__ beta,
                                                 int C, int cpg) {
    int grp = blockIdx.x, b = blockIdx.y, tid = threadIdx.x;
    __shared__ float red[256];
    float* xb = x + (size_t)b * C + (size_t)grp * cpg;
    float v = (tid < cpg) ? xb[tid] : 0.f;
    red[tid] = v;
    __syncthreads();
    for (int s2 = 128; s2 > 0; s2 >>= 1) {
        if (tid < s2) red[tid] += red[tid + s2];
        __syncthreads();
    }
    float mean = red[0] / (float)cpg;
    __syncthreads();
    float d = (tid < cpg) ? (v - mean) : 0.f;
    red[tid] = d * d;
    __syncthreads();
    for (int s2 = 128; s2 > 0; s2 >>= 1) {
        if (tid < s2) red[tid] += red[tid + s2];
        __syncthreads();
    }
    float inv = rsqrtf(red[0] / (float)cpg + 1e-5f);
    if (tid < cpg) {
        int c = grp * cpg + tid;
        float y = (v - mean) * inv * gamma[c] + beta[c];
        xb[tid] = y > 0.f ? y : 0.01f * y;
    }
}

// ---------------------------------------------------------------------------
extern "C" void kernel_launch(void* const* d_in, const int* in_sizes, int n_in,
                              void* d_out, int out_size, void* d_ws, size_t ws_size,
                              hipStream_t stream) {
    (void)in_sizes; (void)n_in; (void)out_size; (void)ws_size;
    // d_in layout (flattened setup_inputs):
    // [0] point_cloud; [1..6] sa1 w1,b1,w2,b2,w3,b3; [7..12] sa2; [13..18] sa3;
    // [19..28] fc: w1,b1,g1,be1,w2,b2,g2,be2,w3,b3
    char* wsb = (char*)d_ws;
    size_t off = 0;
    auto alloc = [&](size_t bytes) -> void* {
        void* p = wsb + off;
        off = (off + bytes + 255) & ~(size_t)255;
        return p;
    };
    _Float16* sa1w1 = (_Float16*)alloc((size_t)64 * 32 * 2);
    _Float16* sa1w2 = (_Float16*)alloc((size_t)64 * 64 * 2);
    _Float16* sa1w3 = (_Float16*)alloc((size_t)64 * 64 * 2);
    _Float16* sa2w1 = (_Float16*)alloc((size_t)128 * 96 * 2);
    _Float16* sa2w2 = (_Float16*)alloc((size_t)128 * 128 * 2);
    _Float16* sa2w3 = (_Float16*)alloc((size_t)256 * 128 * 2);
    _Float16* sa3w1 = (_Float16*)alloc((size_t)512 * 288 * 2);
    _Float16* sa3w2 = (_Float16*)alloc((size_t)512 * 512 * 2);
    _Float16* sa3w3 = (_Float16*)alloc((size_t)1024 * 512 * 2);
    _Float16* fcw1  = (_Float16*)alloc((size_t)4096 * 1024 * 2);
    _Float16* fcw2  = (_Float16*)alloc((size_t)2048 * 4096 * 2);
    _Float16* fcw3  = (_Float16*)alloc((size_t)2048 * 2048 * 2);
    int*   cidx1  = (int*)alloc((size_t)16 * 512 * 4);
    float* nxyz1  = (float*)alloc((size_t)16 * 512 * 3 * 4);
    int*   gidx1  = (int*)alloc((size_t)16 * 512 * 128 * 4);
    float* feats1 = (float*)alloc((size_t)16 * 64 * 512 * 4);
    int*   cidx2  = (int*)alloc((size_t)16 * 128 * 4);
    float* nxyz2  = (float*)alloc((size_t)16 * 128 * 3 * 4);
    int*   gidx2  = (int*)alloc((size_t)16 * 128 * 128 * 4);
    float* feats2 = (float*)alloc((size_t)16 * 256 * 128 * 4);
    float* part3  = (float*)alloc((size_t)4 * 16 * 1024 * 4);
    float* hbuf   = (float*)alloc((size_t)16 * 1024 * 4);
    float* a1     = (float*)alloc((size_t)16 * 4096 * 4);
    float* a2     = (float*)alloc((size_t)16 * 2048 * 4);

    auto cvt = [&](const void* src, _Float16* dst, int O, int K, int Kpad) {
        size_t tot = (size_t)O * Kpad;
        cvt_kernel<<<(unsigned)((tot + 255) / 256), 256, 0, stream>>>(
            (const float*)src, dst, O, K, Kpad);
    };
    cvt(d_in[1],  sa1w1, 64, 4, 32);
    cvt(d_in[3],  sa1w2, 64, 64, 64);
    cvt(d_in[5],  sa1w3, 64, 64, 64);
    cvt(d_in[7],  sa2w1, 128, 67, 96);
    cvt(d_in[9],  sa2w2, 128, 128, 128);
    cvt(d_in[11], sa2w3, 256, 128, 128);
    cvt(d_in[13], sa3w1, 512, 259, 288);
    cvt(d_in[15], sa3w2, 512, 512, 512);
    cvt(d_in[17], sa3w3, 1024, 512, 512);
    cvt(d_in[19], fcw1, 4096, 1024, 1024);
    cvt(d_in[23], fcw2, 2048, 4096, 4096);
    cvt(d_in[27], fcw3, 2048, 2048, 2048);

    const float* pc = (const float*)d_in[0];

    // ---- SA1 ----
    fps_kernel<<<16, 256, 0, stream>>>(pc, 4, 4096, 512, cidx1, nxyz1);
    bq_kernel<<<(16 * 512) / 8, 256, 0, stream>>>(pc, 4, 4096, nxyz1, 512,
                                                  0.05f * 0.05f, 128, gidx1);
    sa1_kernel<<<dim3(512, 16), 256, 0, stream>>>(
        pc, gidx1, nxyz1, sa1w1, (const float*)d_in[2], sa1w2,
        (const float*)d_in[4], sa1w3, (const float*)d_in[6], feats1);

    // ---- SA2 ----
    fps_kernel<<<16, 256, 0, stream>>>(nxyz1, 3, 512, 128, cidx2, nxyz2);
    bq_kernel<<<(16 * 128) / 8, 256, 0, stream>>>(nxyz1, 3, 512, nxyz2, 128,
                                                  0.3f * 0.3f, 128, gidx2);
    sa2_kernel<<<dim3(128, 16), 256, 0, stream>>>(
        nxyz1, feats1, gidx2, nxyz2, sa2w1, (const float*)d_in[8], sa2w2,
        (const float*)d_in[10], sa2w3, (const float*)d_in[12], feats2);

    // ---- SA3 (group all) ----
    sa3_kernel<<<dim3(4, 16), 256, 0, stream>>>(
        nxyz2, feats2, sa3w1, (const float*)d_in[14], sa3w2,
        (const float*)d_in[16], sa3w3, (const float*)d_in[18], part3);
    maxcomb_kernel<<<(16 * 1024 + 255) / 256, 256, 0, stream>>>(part3, hbuf,
                                                                16 * 1024, 4);

    // ---- FC head ----
    fc_kernel<<<32, 256, 0, stream>>>(hbuf, fcw1, (const float*)d_in[20], a1,
                                      1024, 4096);
    gn_kernel<<<dim3(16, 16), 256, 0, stream>>>(a1, (const float*)d_in[21],
                                                (const float*)d_in[22], 4096, 256);
    fc_kernel<<<16, 256, 0, stream>>>(a1, fcw2, (const float*)d_in[24], a2,
                                      4096, 2048);
    gn_kernel<<<dim3(16, 16), 256, 0, stream>>>(a2, (const float*)d_in[25],
                                                (const float*)d_in[26], 2048, 128);
    fc_kernel<<<16, 256, 0, stream>>>(a2, fcw3, (const float*)d_in[28],
                                      (float*)d_out, 2048, 2048);
}
